// FeatCrossAttnLayer2_30150670418184
// MI455X (gfx1250) — compile-verified, matrix operand order checked by reading, not run on hardware
//
#include <hip/hip_runtime.h>
#include <hip/hip_bf16.h>

#define NCAM 6
#define NH   8
#define NP   4
#define DDIM 256
#define HHT  20
#define WWD  40
#define HWV  800
#define LQ   4800
#define NOFF 384

typedef __attribute__((ext_vector_type(16))) __bf16 v16bf;
typedef __attribute__((ext_vector_type(8)))  float  v8f;

#if __has_builtin(__builtin_amdgcn_sched_barrier)
#define SCHED_FENCE() __builtin_amdgcn_sched_barrier(0)
#else
#define SCHED_FENCE()
#endif

__device__ __forceinline__ unsigned short f2bf(float f) {
    union { float f; unsigned int u; } c; c.f = f;
    unsigned int u = c.u;
    unsigned int r = (u + 0x7FFFu + ((u >> 16) & 1u)) >> 16;
    return (unsigned short)r;
}

union FragBF { v16bf v; uint4 q[2]; };

// ---------------------------------------------------------------------------
// Register-blocked WMMA core: one wave computes a 64x16 f32 block of
// A(MxK,bf16) * W(NxK,bf16)^T as 4 M-subtiles sharing one B fragment per
// K-step. A sched_barrier pins the 10-load batch ahead of the 4 WMMAs so the
// fragments stay in distinct VGPRs and waits are partial (0x6/0x4/0x2/0x0)
// instead of a full s_wait_loadcnt 0x0 per WMMA.
// A-frag (16-bit 16x32 layout): lane L holds row (L&15); lanes<16 carry K
// 0..7 & 16..23 of each 32-block, lanes>=16 carry K 8..15 & 24..31 (hi8).
// B-frag mirrors A-frag loaded from row-major W (col n holds W[n][k]).
// ---------------------------------------------------------------------------
__device__ __forceinline__ void wmma_block64(const unsigned short* __restrict__ A,
                                             const unsigned short* __restrict__ W,
                                             int K, int m0, int n0, int lane, int hi8,
                                             v8f acc[4]) {
    const unsigned short* ap = A + (size_t)(m0 + (lane & 15)) * K;
    const unsigned short* wp = W + (size_t)(n0 + (lane & 15)) * K;
    const size_t mstep = (size_t)16 * K;   // 16 rows per M-subtile
    for (int k0 = 0; k0 < K; k0 += 32) {
        if (k0 + 32 < K) {                 // pull next K-slice toward near cache
            __builtin_prefetch(wp + k0 + 32, 0, 3);
            __builtin_prefetch(ap + k0 + 32, 0, 3);
            __builtin_prefetch(ap + mstep * 2 + k0 + 32, 0, 3);
        }
        FragBF b, a0, a1, a2, a3;
        b.q[0]  = *(const uint4*)(wp + k0 + hi8);
        b.q[1]  = *(const uint4*)(wp + k0 + 16 + hi8);
        a0.q[0] = *(const uint4*)(ap + k0 + hi8);
        a0.q[1] = *(const uint4*)(ap + k0 + 16 + hi8);
        a1.q[0] = *(const uint4*)(ap + mstep + k0 + hi8);
        a1.q[1] = *(const uint4*)(ap + mstep + k0 + 16 + hi8);
        a2.q[0] = *(const uint4*)(ap + 2 * mstep + k0 + hi8);
        a2.q[1] = *(const uint4*)(ap + 2 * mstep + k0 + 16 + hi8);
        a3.q[0] = *(const uint4*)(ap + 3 * mstep + k0 + hi8);
        a3.q[1] = *(const uint4*)(ap + 3 * mstep + k0 + 16 + hi8);
        SCHED_FENCE();                     // keep load batch ahead of WMMA batch
        acc[0] = __builtin_amdgcn_wmma_f32_16x16x32_bf16(false, a0.v, false, b.v,
                                                         (short)0, acc[0], false, false);
        acc[1] = __builtin_amdgcn_wmma_f32_16x16x32_bf16(false, a1.v, false, b.v,
                                                         (short)0, acc[1], false, false);
        acc[2] = __builtin_amdgcn_wmma_f32_16x16x32_bf16(false, a2.v, false, b.v,
                                                         (short)0, acc[2], false, false);
        acc[3] = __builtin_amdgcn_wmma_f32_16x16x32_bf16(false, a3.v, false, b.v,
                                                         (short)0, acc[3], false, false);
    }
}

struct TileInfo { int m0, n0, lane, n, mbase, hi8; };

// Mb = M/64 blocks, Nt = N/16 tiles; nt fastest so waves in a block share A rows
__device__ __forceinline__ bool tile_setup64(int Mb, int Nt, TileInfo& t) {
    int wv   = threadIdx.x >> 5;
    int tile = blockIdx.x * (blockDim.x >> 5) + wv;
    if (tile >= Mb * Nt) return false;           // wave-uniform exit: EXEC stays all-1s
    t.lane  = threadIdx.x & 31;
    int mb  = tile / Nt, nt = tile % Nt;
    t.m0    = mb * 64;
    t.n0    = nt * 16;
    t.n     = t.n0 + (t.lane & 15);
    t.hi8   = (t.lane >> 4) << 3;                // 0 or 8
    t.mbase = t.m0 + t.hi8;                      // C/D: lanes>=16 hold M=r+8
    return true;
}

#define GEMM_PROLOGUE()                                                        \
    TileInfo t; if (!tile_setup64(M / 64, N / 16, t)) return;                  \
    v8f acc[4] = {{0.f,0.f,0.f,0.f,0.f,0.f,0.f,0.f},                           \
                  {0.f,0.f,0.f,0.f,0.f,0.f,0.f,0.f},                           \
                  {0.f,0.f,0.f,0.f,0.f,0.f,0.f,0.f},                           \
                  {0.f,0.f,0.f,0.f,0.f,0.f,0.f,0.f}};                          \
    wmma_block64(A, Wt, K, t.m0, t.n0, t.lane, t.hi8, acc);                    \
    float bn = bias[t.n];

// GEMM1 epilogue: src (bf16) and spc = src + pos + cam_emb (f32 + bf16)
__global__ void gemm_src_spc_kernel(const unsigned short* __restrict__ A,
                                    const unsigned short* __restrict__ Wt,
                                    const float* __restrict__ bias,
                                    const float* __restrict__ pos,
                                    const float* __restrict__ cam,
                                    unsigned short* __restrict__ srcb,
                                    float* __restrict__ spcf,
                                    unsigned short* __restrict__ spcb,
                                    int M, int N, int K) {
    GEMM_PROLOGUE();
#pragma unroll
    for (int s = 0; s < 4; s++)
#pragma unroll
        for (int r = 0; r < 8; r++) {
            int m    = t.mbase + s * 16 + r;
            int camv = m / HWV, hw = m % HWV;
            float v  = acc[s][r] + bn;
            size_t o = (size_t)m * N + t.n;
            srcb[o]  = f2bf(v);
            float sp = v + pos[(size_t)hw * N + t.n] + cam[(size_t)camv * N + t.n];
            spcf[o]  = sp;
            spcb[o]  = f2bf(sp);
        }
}

// plain f32 output
__global__ void gemm_f32_kernel(const unsigned short* __restrict__ A,
                                const unsigned short* __restrict__ Wt,
                                const float* __restrict__ bias,
                                float* __restrict__ out,
                                int M, int N, int K) {
    GEMM_PROLOGUE();
#pragma unroll
    for (int s = 0; s < 4; s++)
#pragma unroll
        for (int r = 0; r < 8; r++)
            out[(size_t)(t.mbase + s * 16 + r) * N + t.n] = acc[s][r] + bn;
}

// keys/value scattered into kv[(cam*8+head)*800+hw][64] with koff 0/32
__global__ void gemm_kv_kernel(const unsigned short* __restrict__ A,
                               const unsigned short* __restrict__ Wt,
                               const float* __restrict__ bias,
                               float* __restrict__ kv, int koff,
                               int M, int N, int K) {
    GEMM_PROLOGUE();
    int head = t.n >> 5, dd = t.n & 31;
#pragma unroll
    for (int s = 0; s < 4; s++)
#pragma unroll
        for (int r = 0; r < 8; r++) {
            int m    = t.mbase + s * 16 + r;
            int camv = m / HWV, hw = m % HWV;
            kv[(((size_t)camv * NH + head) * HWV + hw) * 64 + koff + dd] = acc[s][r] + bn;
        }
}

// bf16 output, optional relu
__global__ void gemm_bf16_kernel(const unsigned short* __restrict__ A,
                                 const unsigned short* __restrict__ Wt,
                                 const float* __restrict__ bias,
                                 unsigned short* __restrict__ out, int relu,
                                 int M, int N, int K) {
    GEMM_PROLOGUE();
#pragma unroll
    for (int s = 0; s < 4; s++)
#pragma unroll
        for (int r = 0; r < 8; r++) {
            float v = acc[s][r] + bn;
            if (relu) v = fmaxf(v, 0.f);
            out[(size_t)(t.mbase + s * 16 + r) * N + t.n] = f2bf(v);
        }
}

// f32 output + elementwise add of another f32 matrix (residual / skip)
__global__ void gemm_addf32_kernel(const unsigned short* __restrict__ A,
                                   const unsigned short* __restrict__ Wt,
                                   const float* __restrict__ bias,
                                   const float* __restrict__ add,
                                   float* __restrict__ out,
                                   int M, int N, int K) {
    GEMM_PROLOGUE();
#pragma unroll
    for (int s = 0; s < 4; s++)
#pragma unroll
        for (int r = 0; r < 8; r++) {
            size_t o = (size_t)(t.mbase + s * 16 + r) * N + t.n;
            out[o]   = acc[s][r] + bn + add[o];
        }
}

// ---------------------------------------------------------------------------
// Helper kernels
// ---------------------------------------------------------------------------
__global__ void prep_kernel(const float* __restrict__ x, float* __restrict__ res,
                            unsigned short* __restrict__ xb) {
    int i = blockIdx.x * blockDim.x + threadIdx.x;
    if (i >= LQ * DDIM) return;
    int q = i / DDIM, c = i % DDIM;
    int cam = q / HWV, hw = q % HWV, h = hw / WWD, w = hw % WWD;
    float v = x[(((size_t)cam * DDIM + c) * HHT + h) * WWD + w];
    res[i] = v;
    xb[i]  = f2bf(v);
}

__global__ void pos_kernel(float* __restrict__ pos) {
    int i = blockIdx.x * blockDim.x + threadIdx.x;
    if (i >= HWV * DDIM) return;
    int hw = i / DDIM, c = i % DDIM;
    int h = hw / WWD, w = hw % WWD;
    const float two_pi = 6.283185307179586f;
    float val;
    if (c < 128) {
        int j = c >> 1;
        float tt  = powf(10000.f, (float)j / 64.f);
        float v   = (float)(h + 1) / (HHT + 1e-6f) * two_pi;
        float ang = v / tt;
        val = (c & 1) ? cosf(ang) : sinf(ang);
    } else {
        int cc = c - 128, j = cc >> 1;
        float tt  = powf(10000.f, (float)j / 64.f);
        float v   = (float)(w + 1) / (WWD + 1e-6f) * two_pi;
        float ang = v / tt;
        val = (cc & 1) ? cosf(ang) : sinf(ang);
    }
    pos[i] = val;
}

__global__ void cvt_kernel(const float* __restrict__ in,
                           unsigned short* __restrict__ out, int n) {
    int i = blockIdx.x * blockDim.x + threadIdx.x;
    if (i < n) out[i] = f2bf(in[i]);
}

__global__ void mean_kernel(const float* __restrict__ t, float* __restrict__ g) {
    int i = blockIdx.x * blockDim.x + threadIdx.x;
    if (i >= NCAM * DDIM) return;
    int cam = i / DDIM, c = i % DDIM;
    float s = 0.f;
    for (int r = 0; r < HWV; r++)
        s += t[((size_t)cam * HWV + r) * DDIM + c];
    g[i] = s * (1.f / HWV);
}

// ---------------------------------------------------------------------------
// Deformable attention: one wave32 per (query, head); lane = head-dim (hd=32).
// 24 bilinear-sampled points + 6 global points; softmax over 30.
// ---------------------------------------------------------------------------
__global__ void attn_kernel(const float* __restrict__ spc,
                            const float* __restrict__ off,
                            const float* __restrict__ kv,
                            const float* __restrict__ g,
                            float* __restrict__ aof,
                            unsigned short* __restrict__ aob) {
    int wid = (blockIdx.x * blockDim.x + threadIdx.x) >> 5;
    if (wid >= LQ * NH) return;
    int lane = threadIdx.x & 31;
    int q = wid >> 3, head = wid & 7;
    int hw = q % HWV, h = hw / WWD, w = hw % WWD;
    float rx = (w + 0.5f) / (float)WWD;
    float ry = (h + 0.5f) / (float)HHT;
    float qd = spc[(size_t)q * DDIM + head * 32 + lane];
    const float scale = 0.17677669529663687f; // 1/sqrt(32)

    float logits[30], vvals[30];
#pragma unroll
    for (int l = 0; l < NCAM; l++) {
        const float* kvb = kv + (((size_t)l * NH + head) * HWV) * 64;
#pragma unroll
        for (int p = 0; p < NP; p++) {
            size_t ob = (size_t)q * NOFF + (((size_t)head * NCAM + l) * NP + p) * 2;
            float ox = off[ob + 0], oy = off[ob + 1];
            float xf = (rx + ox / (float)WWD) * (float)WWD - 0.5f;
            float yf = (ry + oy / (float)HHT) * (float)HHT - 0.5f;
            float x0 = floorf(xf), y0 = floorf(yf);
            float fx = xf - x0, fy = yf - y0;
            float kacc = 0.f, vacc = 0.f;
#pragma unroll
            for (int dx = 0; dx < 2; dx++)
#pragma unroll
                for (int dy = 0; dy < 2; dy++) {
                    float xi = x0 + dx, yi = y0 + dy;
                    if (xi >= 0.f && xi < (float)WWD && yi >= 0.f && yi < (float)HHT) {
                        float wgt = (dx ? fx : 1.f - fx) * (dy ? fy : 1.f - fy);
                        const float* cell = kvb + ((size_t)((int)yi * WWD + (int)xi)) * 64;
                        kacc += wgt * cell[lane];
                        vacc += wgt * cell[32 + lane];
                    }
                }
            float dot = qd * kacc;
#pragma unroll
            for (int s = 16; s > 0; s >>= 1) dot += __shfl_xor(dot, s, 32);
            logits[l * NP + p] = dot * scale;
            vvals[l * NP + p]  = vacc;
        }
    }
#pragma unroll
    for (int l = 0; l < NCAM; l++) {
        float gd  = g[(size_t)l * DDIM + head * 32 + lane];
        float dot = qd * gd;
#pragma unroll
        for (int s = 16; s > 0; s >>= 1) dot += __shfl_xor(dot, s, 32);
        logits[24 + l] = dot * scale;
        vvals[24 + l]  = gd;
    }
    float mx = logits[0];
#pragma unroll
    for (int i = 1; i < 30; i++) mx = fmaxf(mx, logits[i]);
    float se = 0.f, outd = 0.f;
#pragma unroll
    for (int i = 0; i < 30; i++) {
        float e = expf(logits[i] - mx);
        se   += e;
        outd += e * vvals[i];
    }
    outd /= se;
    size_t o = (size_t)q * DDIM + head * 32 + lane;
    aof[o] = outd;
    aob[o] = f2bf(outd);
}

// ---------------------------------------------------------------------------
// LayerNorm, block per row (256 threads)
// ---------------------------------------------------------------------------
__global__ void ln_kernel(const float* __restrict__ xin, const float* __restrict__ gw,
                          const float* __restrict__ bw, float* __restrict__ outf,
                          unsigned short* __restrict__ outb) {
    __shared__ float red[DDIM];
    int row = blockIdx.x, c = threadIdx.x;
    float v = xin[(size_t)row * DDIM + c];
    red[c] = v; __syncthreads();
    for (int s = 128; s > 0; s >>= 1) { if (c < s) red[c] += red[c + s]; __syncthreads(); }
    float mu = red[0] * (1.f / DDIM);
    __syncthreads();
    float d = v - mu;
    red[c] = d * d; __syncthreads();
    for (int s = 128; s > 0; s >>= 1) { if (c < s) red[c] += red[c + s]; __syncthreads(); }
    float var = red[0] * (1.f / DDIM);
    float y = d * rsqrtf(var + 1e-5f) * gw[c] + bw[c];
    outf[(size_t)row * DDIM + c] = y;
    outb[(size_t)row * DDIM + c] = f2bf(y);
}

// final LayerNorm fused with NHWC -> NCHW output transpose
__global__ void ln_out_kernel(const float* __restrict__ xin, const float* __restrict__ gw,
                              const float* __restrict__ bw, float* __restrict__ out) {
    __shared__ float red[DDIM];
    int row = blockIdx.x, c = threadIdx.x;
    float v = xin[(size_t)row * DDIM + c];
    red[c] = v; __syncthreads();
    for (int s = 128; s > 0; s >>= 1) { if (c < s) red[c] += red[c + s]; __syncthreads(); }
    float mu = red[0] * (1.f / DDIM);
    __syncthreads();
    float d = v - mu;
    red[c] = d * d; __syncthreads();
    for (int s = 128; s > 0; s >>= 1) { if (c < s) red[c] += red[c + s]; __syncthreads(); }
    float var = red[0] * (1.f / DDIM);
    float y = d * rsqrtf(var + 1e-5f) * gw[c] + bw[c];
    int cam = row / HWV, hw = row % HWV, h = hw / WWD, w = hw % WWD;
    out[(((size_t)cam * DDIM + c) * HHT + h) * WWD + w] = y;
}

// ---------------------------------------------------------------------------
extern "C" void kernel_launch(void* const* d_in, const int* in_sizes, int n_in,
                              void* d_out, int out_size, void* d_ws, size_t ws_size,
                              hipStream_t stream) {
    (void)in_sizes; (void)n_in; (void)out_size; (void)ws_size;
    const float* x      = (const float*)d_in[0];
    const float* conv0w = (const float*)d_in[1];
    const float* conv0b = (const float*)d_in[2];
    const float* conv2w = (const float*)d_in[3];
    const float* conv2b = (const float*)d_in[4];
    const float* camemb = (const float*)d_in[5];
    const float* offw   = (const float*)d_in[6];
    const float* offb   = (const float*)d_in[7];
    const float* valw   = (const float*)d_in[8];
    const float* valb   = (const float*)d_in[9];
    const float* keyw   = (const float*)d_in[10];
    const float* keyb   = (const float*)d_in[11];
    const float* outw   = (const float*)d_in[12];
    const float* outb   = (const float*)d_in[13];
    const float* conv1w = (const float*)d_in[14];
    const float* conv1b = (const float*)d_in[15];
    const float* n0g    = (const float*)d_in[16];
    const float* n0b    = (const float*)d_in[17];
    const float* lin1w  = (const float*)d_in[18];
    const float* lin1b  = (const float*)d_in[19];
    const float* lin2w  = (const float*)d_in[20];
    const float* lin2b  = (const float*)d_in[21];
    const float* n2g    = (const float*)d_in[22];
    const float* n2b    = (const float*)d_in[23];

    char* ws = (char*)d_ws;
    size_t cur = 0;
    auto alloc = [&](size_t b) { size_t r = cur; cur += (b + 255) & ~(size_t)255; return r; };

    float* res    = (float*)(ws + alloc((size_t)LQ * DDIM * 4));
    float* spcf   = (float*)(ws + alloc((size_t)LQ * DDIM * 4));
    float* kvbuf  = (float*)(ws + alloc((size_t)NCAM * NH * HWV * 64 * 4));
    float* gbuf   = (float*)(ws + alloc((size_t)NCAM * DDIM * 4));
    float* buf1   = (float*)(ws + alloc((size_t)LQ * NOFF * 4));   // offsets, then y2
    float* buf2   = (float*)(ws + alloc((size_t)LQ * DDIM * 4));   // conv2 tmp, then y1
    float* out0f  = (float*)(ws + alloc((size_t)LQ * DDIM * 4));
    float* aof    = (float*)(ws + alloc((size_t)LQ * DDIM * 4));
    float* pos    = (float*)(ws + alloc((size_t)HWV * DDIM * 4));
    unsigned short* xb   = (unsigned short*)(ws + alloc((size_t)LQ * DDIM * 2)); // x, then ao2
    unsigned short* srcb = (unsigned short*)(ws + alloc((size_t)LQ * DDIM * 2)); // src, then h1
    unsigned short* spcb = (unsigned short*)(ws + alloc((size_t)LQ * DDIM * 2)); // spc, then out0
    unsigned short* aob  = (unsigned short*)(ws + alloc((size_t)LQ * DDIM * 2));
    unsigned short* wb_conv0 = (unsigned short*)(ws + alloc(65536 * 2));
    unsigned short* wb_conv2 = (unsigned short*)(ws + alloc(65536 * 2));
    unsigned short* wb_val   = (unsigned short*)(ws + alloc(65536 * 2));
    unsigned short* wb_key   = (unsigned short*)(ws + alloc(65536 * 2));
    unsigned short* wb_off   = (unsigned short*)(ws + alloc(98304 * 2));
    unsigned short* wb_out   = (unsigned short*)(ws + alloc(65536 * 2));
    unsigned short* wb_conv1 = (unsigned short*)(ws + alloc(65536 * 2));
    unsigned short* wb_lin1  = (unsigned short*)(ws + alloc(65536 * 2));
    unsigned short* wb_lin2  = (unsigned short*)(ws + alloc(65536 * 2));
    unsigned short* ao2b  = xb;    // reuse (x consumed by GEMM1)
    unsigned short* h1b   = srcb;  // reuse (src consumed by key/val GEMMs)
    unsigned short* out0b = spcb;  // reuse (spc consumed by offsets GEMM)
    float* y1 = buf2;
    float* y2 = buf1;

    dim3 B(256);
    prep_kernel<<<dim3(LQ), B, 0, stream>>>(x, res, xb);
    pos_kernel<<<dim3(HWV), B, 0, stream>>>(pos);
    cvt_kernel<<<dim3(256), B, 0, stream>>>(conv0w, wb_conv0, 65536);
    cvt_kernel<<<dim3(256), B, 0, stream>>>(conv2w, wb_conv2, 65536);
    cvt_kernel<<<dim3(256), B, 0, stream>>>(valw,   wb_val,   65536);
    cvt_kernel<<<dim3(256), B, 0, stream>>>(keyw,   wb_key,   65536);
    cvt_kernel<<<dim3(384), B, 0, stream>>>(offw,   wb_off,   98304);
    cvt_kernel<<<dim3(256), B, 0, stream>>>(outw,   wb_out,   65536);
    cvt_kernel<<<dim3(256), B, 0, stream>>>(conv1w, wb_conv1, 65536);
    cvt_kernel<<<dim3(256), B, 0, stream>>>(lin1w,  wb_lin1,  65536);
    cvt_kernel<<<dim3(256), B, 0, stream>>>(lin2w,  wb_lin2,  65536);

    // 64x16 blocks: N=256 -> 75*16=1200 wave-tiles = 150 blocks; N=384 -> 225 blocks
    gemm_src_spc_kernel<<<dim3(150), B, 0, stream>>>(xb, wb_conv0, conv0b, pos, camemb,
                                                     srcb, spcf, spcb, LQ, DDIM, DDIM);
    gemm_f32_kernel<<<dim3(150), B, 0, stream>>>(spcb, wb_conv2, conv2b, buf2, LQ, DDIM, DDIM);
    mean_kernel<<<dim3(6), B, 0, stream>>>(buf2, gbuf);
    gemm_kv_kernel<<<dim3(150), B, 0, stream>>>(srcb, wb_key, keyb, kvbuf, 0,  LQ, DDIM, DDIM);
    gemm_kv_kernel<<<dim3(150), B, 0, stream>>>(srcb, wb_val, valb, kvbuf, 32, LQ, DDIM, DDIM);
    gemm_f32_kernel<<<dim3(225), B, 0, stream>>>(spcb, wb_off, offb, buf1, LQ, NOFF, DDIM);
    attn_kernel<<<dim3(LQ), B, 0, stream>>>(spcf, buf1, kvbuf, gbuf, aof, aob);
    gemm_bf16_kernel<<<dim3(150), B, 0, stream>>>(aob, wb_out, outb, ao2b, 0, LQ, DDIM, DDIM);
    gemm_addf32_kernel<<<dim3(150), B, 0, stream>>>(ao2b, wb_conv1, conv1b, res, y1, LQ, DDIM, DDIM);
    ln_kernel<<<dim3(LQ), B, 0, stream>>>(y1, n0g, n0b, out0f, out0b);
    gemm_bf16_kernel<<<dim3(150), B, 0, stream>>>(out0b, wb_lin1, lin1b, h1b, 1, LQ, DDIM, DDIM);
    gemm_addf32_kernel<<<dim3(150), B, 0, stream>>>(h1b, wb_lin2, lin2b, out0f, y2, LQ, DDIM, DDIM);
    ln_out_kernel<<<dim3(LQ), B, 0, stream>>>(y2, n2g, n2b, (float*)d_out);
}
